// GatedUpdate_49924699848927
// MI455X (gfx1250) — matrix operand order repaired and checked by reading, not compiled, not run on hardware
//
#include <hip/hip_runtime.h>
#include <hip/hip_bf16.h>

// ---------------------------------------------------------------------------
// GatedUpdate for MI455X (gfx1250, wave32, WMMA + async-to-LDS).
// Memory-bound (~152 MB fp32 weights @ 23.3 TB/s => ~6.5us floor).
// fp32 weights converted to bf16 in-register, fed to v_wmma_f32_16x16x32_bf16.
// A-operand tiles are DMA'd into LDS with global_load_async_to_lds_b128
// (ASYNCcnt), double-buffered against the WMMA loop.
// ---------------------------------------------------------------------------

typedef __attribute__((ext_vector_type(16))) __bf16 v16bf;
typedef __attribute__((ext_vector_type(8)))  float  v8f;

#define BB   32
#define NN   64
#define DD   384
#define K1   1152   // 3*D
#define KG   768    // 2*D
#define ROWS 2048   // B*N

union Frag { uint4 q[2]; v16bf v; };

__device__ __forceinline__ unsigned short f2bf(float f) {
  unsigned u = __float_as_uint(f);
  u += 0x7FFFu + ((u >> 16) & 1u);          // round-to-nearest-even
  return (unsigned short)(u >> 16);
}

__device__ __forceinline__ float gelu_exact(float x) {
  return 0.5f * x * (1.0f + erff(x * 0.7071067811865475f));
}

__device__ __forceinline__ float wave_sum(float v) {
  for (int off = 16; off > 0; off >>= 1) v += __shfl_xor(v, off, 32);
  return v;
}

// --- CDNA5 async copy: global -> LDS, tracked by ASYNCcnt ------------------
__device__ __forceinline__ void async_b128(unsigned lds_off, const void* gptr) {
  asm volatile("global_load_async_to_lds_b128 %0, %1, off"
               :: "v"(lds_off), "v"((unsigned long long)(uintptr_t)gptr)
               : "memory");
}
#define S_WAIT_ASYNC_1() asm volatile("s_wait_asynccnt 0x1" ::: "memory")
#define S_WAIT_ASYNC_0() asm volatile("s_wait_asynccnt 0x0" ::: "memory")

// Copy a 32-row x 32-k bf16 A tile (64 B/row) into LDS buffer.
// Threads 0..127 (waves 0..3) each issue exactly one b128 async transfer.
__device__ __forceinline__ void issue_A_async(const unsigned short* Abase,
                                              int strideK, int k0,
                                              unsigned short (*Alds)[32], int tid) {
  if (tid < 128) {
    int r = tid >> 2, c = tid & 3;
    const void* g = Abase + (size_t)r * strideK + k0 + c * 8;
    unsigned loff = (unsigned)(uintptr_t)(&Alds[r][c * 8]);
    async_b128(loff, g);
  }
}

// ---------------------------------------------------------------------------
// Prep kernels
// ---------------------------------------------------------------------------
__global__ void k_norm(const float* __restrict__ update, float* __restrict__ unorm) {
  int b = blockIdx.x, lane = threadIdx.x;
  float s = 0.f;
  for (int i = lane; i < DD; i += 32) { float v = update[b * DD + i]; s += v * v; }
  s = wave_sum(s);
  if (lane == 0) unorm[b] = sqrtf(s);
}

__global__ void k_mean(const float* __restrict__ memory, float* __restrict__ memmean) {
  int idx = blockIdx.x * 256 + threadIdx.x;   // 0 .. 32*384-1
  int b = idx / DD, d = idx % DD;
  float s = 0.f;
  for (int n = 0; n < NN; ++n) s += memory[((size_t)b * NN + n) * DD + d];
  memmean[idx] = s * (1.0f / NN);
}

// builds bf16 A-operand x = [update, mem*||u||, mean(mem)*||u||] row=(n*32+b)
// and the mem_scaled half of the gate input.
__global__ void k_build(const float* __restrict__ memory, const float* __restrict__ update,
                        const float* __restrict__ unorm, const float* __restrict__ memmean,
                        unsigned short* __restrict__ xbf, unsigned short* __restrict__ ginbf) {
  int idx = blockIdx.x * 256 + threadIdx.x;   // n*32*384 + b*384 + d
  int d = idx % DD;
  int b = (idx / DD) & 31;
  int n = idx / (DD * 32);
  float us = unorm[b];
  float ms = memory[((size_t)b * NN + n) * DD + d] * us;
  int row = n * 32 + b;
  unsigned short* xr = xbf + (size_t)row * K1;
  xr[d]           = f2bf(update[b * DD + d]);
  xr[DD + d]      = f2bf(ms);
  xr[2 * DD + d]  = f2bf(memmean[b * DD + d] * us);
  ginbf[(size_t)row * KG + DD + d] = f2bf(ms);
}

// ---------------------------------------------------------------------------
// GEMM1: per token n, h = gelu(x(32x1152) @ W1[n](1152x384) + b1[n]) -> bf16
// block = 256 thr (8 waves); grid = (3 col-tiles of 128, 64 tokens)
// ---------------------------------------------------------------------------
__global__ void k_gemm1(const unsigned short* __restrict__ xbf,
                        const float* __restrict__ W1, const float* __restrict__ b1,
                        unsigned short* __restrict__ hbf) {
  __shared__ alignas(32) unsigned short Wlds[128][32];      // [d_local][k] bf16
  __shared__ alignas(16) unsigned short Alds[2][32][32];    // double-buffered A
  int n = blockIdx.y;
  int dbase = blockIdx.x * 128;
  int tid = threadIdx.x;
  int wave = tid >> 5, lane = tid & 31;
  int g = lane >> 4, nl = lane & 15;
  int col_local = wave * 16 + nl;
  int col = dbase + col_local;

  __builtin_amdgcn_s_wait_tensorcnt((short)0);   // no-op safe; TENSORcnt==0

  const float* Wg = W1 + (size_t)n * K1 * DD;
  const unsigned short* Abase = xbf + (size_t)(n * 32) * K1;

  issue_A_async(Abase, K1, 0, Alds[0], tid);

  v8f c0 = {}; v8f c1 = {};
  for (int k0 = 0; k0 < K1; k0 += 32) {
    int buf = (k0 >> 5) & 1;
    // stage 32x128 fp32 W tile -> bf16 LDS, transposed to [d][k]
    for (int i = 0; i < 4; ++i) {
      int linear = tid + i * 256;
      int kk = linear >> 5, dq = linear & 31;
      const float4 w = *(const float4*)(Wg + (size_t)(k0 + kk) * DD + dbase + dq * 4);
      Wlds[dq * 4 + 0][kk] = f2bf(w.x);
      Wlds[dq * 4 + 1][kk] = f2bf(w.y);
      Wlds[dq * 4 + 2][kk] = f2bf(w.z);
      Wlds[dq * 4 + 3][kk] = f2bf(w.w);
    }
    if (k0 + 32 < K1) {
      __builtin_prefetch(Wg + (size_t)(k0 + 32) * DD + dbase + (tid & 31) * 4, 0, 1);
      issue_A_async(Abase, K1, k0 + 32, Alds[buf ^ 1], tid);
      S_WAIT_ASYNC_1();          // current buffer's DMA done (in-order)
    } else {
      S_WAIT_ASYNC_0();
    }
    __syncthreads();

    const unsigned short* ar0 = &Alds[buf][nl][g * 8];
    const unsigned short* ar1 = &Alds[buf][16 + nl][g * 8];
    Frag a0, a1, bf;
    a0.q[0] = *(const uint4*)(ar0);
    a0.q[1] = *(const uint4*)(ar0 + 16);
    a1.q[0] = *(const uint4*)(ar1);
    a1.q[1] = *(const uint4*)(ar1 + 16);
    bf.q[0] = *(const uint4*)(&Wlds[col_local][g * 8]);
    bf.q[1] = *(const uint4*)(&Wlds[col_local][16 + g * 8]);

    c0 = __builtin_amdgcn_wmma_f32_16x16x32_bf16(false, a0.v, false, bf.v, (short)0, c0, false, false);
    c1 = __builtin_amdgcn_wmma_f32_16x16x32_bf16(false, a1.v, false, bf.v, (short)0, c1, false, false);
    __syncthreads();
  }

  float bias = b1[n * DD + col];
  for (int r = 0; r < 8; ++r) {
    int m0 = g * 8 + r;
    hbf[((size_t)(n * 32) + m0) * DD + col] = f2bf(gelu_exact(c0[r] + bias));
    int m1 = 16 + g * 8 + r;
    hbf[((size_t)(n * 32) + m1) * DD + col] = f2bf(gelu_exact(c1[r] + bias));
  }
}

// ---------------------------------------------------------------------------
// GEMM2: deltas = h(32x384) @ W2[n](384x384) + b2[n]; emits
// delta_diff fp32 and bf16 half of gate input.
// ---------------------------------------------------------------------------
__global__ void k_gemm2(const unsigned short* __restrict__ hbf,
                        const float* __restrict__ W2, const float* __restrict__ b2,
                        const float* __restrict__ memory,
                        float* __restrict__ ddws, unsigned short* __restrict__ ginbf) {
  __shared__ alignas(32) unsigned short Wlds[128][32];
  __shared__ alignas(16) unsigned short Alds[2][32][32];
  int n = blockIdx.y;
  int dbase = blockIdx.x * 128;
  int tid = threadIdx.x;
  int wave = tid >> 5, lane = tid & 31;
  int g = lane >> 4, nl = lane & 15;
  int col_local = wave * 16 + nl;
  int col = dbase + col_local;

  const float* Wg = W2 + (size_t)n * DD * DD;
  const unsigned short* Abase = hbf + (size_t)(n * 32) * DD;

  issue_A_async(Abase, DD, 0, Alds[0], tid);

  v8f c0 = {}; v8f c1 = {};
  for (int k0 = 0; k0 < DD; k0 += 32) {
    int buf = (k0 >> 5) & 1;
    for (int i = 0; i < 4; ++i) {
      int linear = tid + i * 256;
      int kk = linear >> 5, dq = linear & 31;
      const float4 w = *(const float4*)(Wg + (size_t)(k0 + kk) * DD + dbase + dq * 4);
      Wlds[dq * 4 + 0][kk] = f2bf(w.x);
      Wlds[dq * 4 + 1][kk] = f2bf(w.y);
      Wlds[dq * 4 + 2][kk] = f2bf(w.z);
      Wlds[dq * 4 + 3][kk] = f2bf(w.w);
    }
    if (k0 + 32 < DD) {
      issue_A_async(Abase, DD, k0 + 32, Alds[buf ^ 1], tid);
      S_WAIT_ASYNC_1();
    } else {
      S_WAIT_ASYNC_0();
    }
    __syncthreads();

    const unsigned short* ar0 = &Alds[buf][nl][g * 8];
    const unsigned short* ar1 = &Alds[buf][16 + nl][g * 8];
    Frag a0, a1, bf;
    a0.q[0] = *(const uint4*)(ar0);
    a0.q[1] = *(const uint4*)(ar0 + 16);
    a1.q[0] = *(const uint4*)(ar1);
    a1.q[1] = *(const uint4*)(ar1 + 16);
    bf.q[0] = *(const uint4*)(&Wlds[col_local][g * 8]);
    bf.q[1] = *(const uint4*)(&Wlds[col_local][16 + g * 8]);

    c0 = __builtin_amdgcn_wmma_f32_16x16x32_bf16(false, a0.v, false, bf.v, (short)0, c0, false, false);
    c1 = __builtin_amdgcn_wmma_f32_16x16x32_bf16(false, a1.v, false, bf.v, (short)0, c1, false, false);
    __syncthreads();
  }

  float bias = b2[n * DD + col];
  for (int r = 0; r < 8; ++r) {
    int m0 = g * 8 + r;
    float dd0 = c0[r] + bias - memory[((size_t)m0 * NN + n) * DD + col];
    int row0 = n * 32 + m0;
    ddws[(size_t)row0 * DD + col] = dd0;
    ginbf[(size_t)row0 * KG + col] = f2bf(dd0);

    int m1 = 16 + g * 8 + r;
    float dd1 = c1[r] + bias - memory[((size_t)m1 * NN + n) * DD + col];
    int row1 = n * 32 + m1;
    ddws[(size_t)row1 * DD + col] = dd1;
    ginbf[(size_t)row1 * KG + col] = f2bf(dd1);
  }
}

// ---------------------------------------------------------------------------
// Gate: gh = gelu(gate_in(2048x768) @ Wg1(768x384)+bg1); gate = sigmoid(gh@Wg2+bg2)
// block handles 32 rows; fuses the Wg2 dot via LDS ds_add_f32.
// ---------------------------------------------------------------------------
__global__ void k_gate(const unsigned short* __restrict__ ginbf,
                       const float* __restrict__ Wg1, const float* __restrict__ bg1,
                       const float* __restrict__ Wg2, const float* __restrict__ bg2,
                       float* __restrict__ gate) {
  __shared__ alignas(32) unsigned short Wlds[128][32];
  __shared__ alignas(16) unsigned short Alds[2][32][32];
  __shared__ float gacc[32];
  int row0 = blockIdx.x * 32;
  int tid = threadIdx.x;
  int wave = tid >> 5, lane = tid & 31;
  int g = lane >> 4, nl = lane & 15;
  if (tid < 32) gacc[tid] = 0.f;
  __syncthreads();

  const unsigned short* Abase = ginbf + (size_t)row0 * KG;

  for (int p = 0; p < 3; ++p) {
    int cbase = p * 128;
    int col_local = wave * 16 + nl;
    int col = cbase + col_local;
    v8f c0 = {}; v8f c1 = {};

    issue_A_async(Abase, KG, 0, Alds[0], tid);

    for (int k0 = 0; k0 < KG; k0 += 32) {
      int buf = (k0 >> 5) & 1;
      for (int i = 0; i < 4; ++i) {
        int linear = tid + i * 256;
        int kk = linear >> 5, dq = linear & 31;
        const float4 w = *(const float4*)(Wg1 + (size_t)(k0 + kk) * DD + cbase + dq * 4);
        Wlds[dq * 4 + 0][kk] = f2bf(w.x);
        Wlds[dq * 4 + 1][kk] = f2bf(w.y);
        Wlds[dq * 4 + 2][kk] = f2bf(w.z);
        Wlds[dq * 4 + 3][kk] = f2bf(w.w);
      }
      if (k0 + 32 < KG) {
        issue_A_async(Abase, KG, k0 + 32, Alds[buf ^ 1], tid);
        S_WAIT_ASYNC_1();
      } else {
        S_WAIT_ASYNC_0();
      }
      __syncthreads();

      const unsigned short* ar0 = &Alds[buf][nl][g * 8];
      const unsigned short* ar1 = &Alds[buf][16 + nl][g * 8];
      Frag a0, a1, bf;
      a0.q[0] = *(const uint4*)(ar0);
      a0.q[1] = *(const uint4*)(ar0 + 16);
      a1.q[0] = *(const uint4*)(ar1);
      a1.q[1] = *(const uint4*)(ar1 + 16);
      bf.q[0] = *(const uint4*)(&Wlds[col_local][g * 8]);
      bf.q[1] = *(const uint4*)(&Wlds[col_local][16 + g * 8]);

      c0 = __builtin_amdgcn_wmma_f32_16x16x32_bf16(false, a0.v, false, bf.v, (short)0, c0, false, false);
      c1 = __builtin_amdgcn_wmma_f32_16x16x32_bf16(false, a1.v, false, bf.v, (short)0, c1, false, false);
      __syncthreads();
    }
    float bias = bg1[col];
    float w2 = Wg2[col];
    for (int r = 0; r < 8; ++r) {
      atomicAdd(&gacc[g * 8 + r],      gelu_exact(c0[r] + bias) * w2);
      atomicAdd(&gacc[16 + g * 8 + r], gelu_exact(c1[r] + bias) * w2);
    }
    __syncthreads();
  }

  if (tid < 32) {
    float z = gacc[tid] + bg2[0];
    gate[row0 + tid] = 1.0f / (1.0f + expf(-z));
  }
}

// ---------------------------------------------------------------------------
// Final: orthogonalize, normalize, blend, normalize. One wave per (b,n) row.
// ---------------------------------------------------------------------------
__global__ void k_final(const float* __restrict__ memory, const float* __restrict__ ddws,
                        const float* __restrict__ gate, float* __restrict__ out) {
  int tid = threadIdx.x, wave = tid >> 5, lane = tid & 31;
  int row = blockIdx.x * 8 + wave;       // row = n*32 + b
  int n = row >> 5, b = row & 31;
  const float* mrow = memory + ((size_t)b * NN + n) * DD;
  const float* drow = ddws + (size_t)row * DD;

  float ml[12], dl[12];
  float dot = 0.f;
  for (int i = 0; i < 12; ++i) {
    int d = lane + i * 32;
    ml[i] = mrow[d]; dl[i] = drow[d];
    dot += ml[i] * dl[i];
  }
  dot = wave_sum(dot);

  float ss = 0.f;
  for (int i = 0; i < 12; ++i) {
    dl[i] = dl[i] - dot * ml[i];         // delta_orth
    ss += dl[i] * dl[i];
  }
  ss = wave_sum(ss);
  float inv = 1.0f / fmaxf(sqrtf(ss), 1e-12f);
  float gt = gate[row];

  float s2 = 0.f;
  for (int i = 0; i < 12; ++i) {
    dl[i] = ml[i] + gt * dl[i] * inv;
    s2 += dl[i] * dl[i];
  }
  s2 = wave_sum(s2);
  float inv2 = 1.0f / fmaxf(sqrtf(s2), 1e-12f);

  float* orow = out + ((size_t)b * NN + n) * DD;
  for (int i = 0; i < 12; ++i) orow[lane + i * 32] = dl[i] * inv2;
}

// ---------------------------------------------------------------------------
extern "C" void kernel_launch(void* const* d_in, const int* in_sizes, int n_in,
                              void* d_out, int out_size, void* d_ws, size_t ws_size,
                              hipStream_t stream) {
  (void)in_sizes; (void)n_in; (void)out_size; (void)ws_size;
  const float* memory = (const float*)d_in[0];
  const float* update = (const float*)d_in[1];
  const float* W1  = (const float*)d_in[2];
  const float* b1  = (const float*)d_in[3];
  const float* W2  = (const float*)d_in[4];
  const float* b2  = (const float*)d_in[5];
  const float* Wg1 = (const float*)d_in[6];
  const float* bg1 = (const float*)d_in[7];
  const float* Wg2 = (const float*)d_in[8];
  const float* bg2 = (const float*)d_in[9];
  float* out = (float*)d_out;

  char* w = (char*)d_ws;
  float* unorm = (float*)w;            w += 256;
  float* memmean = (float*)w;          w += (size_t)BB * DD * sizeof(float);
  unsigned short* xbf = (unsigned short*)w;   w += (size_t)ROWS * K1 * 2;
  unsigned short* hbf = (unsigned short*)w;   w += (size_t)ROWS * DD * 2;
  unsigned short* ginbf = (unsigned short*)w; w += (size_t)ROWS * KG * 2;
  float* ddws = (float*)w;             w += (size_t)ROWS * DD * sizeof(float);
  float* gate = (float*)w;             w += (size_t)ROWS * sizeof(float);

  k_norm <<<32, 32, 0, stream>>>(update, unorm);
  k_mean <<<48, 256, 0, stream>>>(memory, memmean);
  k_build<<<3072, 256, 0, stream>>>(memory, update, unorm, memmean, xbf, ginbf);
  k_gemm1<<<dim3(3, 64), 256, 0, stream>>>(xbf, W1, b1, hbf);
  k_gemm2<<<dim3(3, 64), 256, 0, stream>>>(hbf, W2, b2, memory, ddws, ginbf);
  k_gate <<<64, 256, 0, stream>>>(ginbf, Wg1, bg1, Wg2, bg2, gate);
  k_final<<<256, 256, 0, stream>>>(memory, ddws, gate, out);
}
